// DeepAR_67284957659733
// MI455X (gfx1250) — compile-verified, matrix operand order checked by reading, not run on hardware
//
#include <hip/hip_runtime.h>

// ---------------------------------------------------------------------------
// DeepAR 3-layer LSTM, B=512, T=128, HID=512, IN=69 (padded to 96)
// v3: 16 blocks x 32 batch rows (M=32 -> each WMMA B-tile reused by 2 M-tiles,
// halving L2 weight traffic vs M=16). h state in LDS (bf16, WMMA A-packed),
// c state in d_ws (f32, L2-resident). Weights pre-packed to bf16 WMMA tiles.
// Gate nonlinearities use hardware rcp/tanh (no IEEE div sequences) so the
// VALU work stays hidden under the WMMA pipe.
// ---------------------------------------------------------------------------

typedef __attribute__((ext_vector_type(16))) __bf16 v16bf;
typedef __attribute__((ext_vector_type(8)))  float  v8f;

#define TSTEPS   128
#define BATCH    512
#define HID      512
#define BQ       32      // batch rows per block (2 M-tiles of 16)
#define NBLOCKS  (BATCH / BQ)   // 16
#define KT_IH0   3       // layer0 ih: K=69 padded to 96 -> 3 k-tiles of 32
#define KT_HH    16      // K=512 -> 16 k-tiles of 32
#define LOG2PI   1.8378770664093453f

__device__ __forceinline__ unsigned short f2bf(float x) {
    union { float f; unsigned int u; } c; c.f = x;
    unsigned int u = c.u;
    return (unsigned short)((u + 0x7FFFu + ((u >> 16) & 1u)) >> 16);
}
__device__ __forceinline__ float bf2f(unsigned short h) {
    union { unsigned int u; float f; } c; c.u = ((unsigned int)h) << 16;
    return c.f;
}
// fast nonlinearities: single v_exp_f32 / v_rcp_f32 (or v_tanh_f32), no IEEE div
__device__ __forceinline__ float sigmoidf_(float x) {
    return __builtin_amdgcn_rcpf(1.f + __expf(-x));
}
__device__ __forceinline__ float tanhf_(float x) {
#if __has_builtin(__builtin_amdgcn_tanhf)
    return __builtin_amdgcn_tanhf(x);
#else
    return 1.f - 2.f * __builtin_amdgcn_rcpf(__expf(2.f * x) + 1.f);
#endif
}
__device__ __forceinline__ v8f wmma_bf16(v16bf a, v16bf b, v8f c) {
    // D = A(16x32 bf16) x B(32x16 bf16) + C(16x16 f32)
    return __builtin_amdgcn_wmma_f32_16x16x32_bf16(
        /*neg_a=*/false, a, /*neg_b=*/false, b,
        /*c_mod=*/(short)0, c, /*reuse_a=*/false, /*reuse_b=*/false);
}

// ---------------------------------------------------------------------------
// Prep: pack a (2048 x Kin) row-major f32 weight into bf16 WMMA B-tiles.
// Packed element e = (ntile*Ktiles + kt)*512 + lane*16 + j, where within a
// 32(K)x16(N) tile: n = lane&15, k = (j&7) + ((j>>3)&1)*16 + ((lane>>4)&1)*8
// (mirrors the ISA 16-bit operand striping). Columns >= Kin are zero padding.
// ---------------------------------------------------------------------------
__global__ void pack_weights_kernel(const float* __restrict__ W,
                                    unsigned short* __restrict__ out,
                                    int Kin, int Ktiles) {
    int total = 128 * Ktiles * 512;   // 128 n-tiles cover 2048 rows
    for (int e = blockIdx.x * blockDim.x + threadIdx.x; e < total;
         e += gridDim.x * blockDim.x) {
        int j     = e & 15;
        int l     = (e >> 4) & 31;
        int tile  = e >> 9;
        int kt    = tile % Ktiles;
        int ntile = tile / Ktiles;
        int n = ntile * 16 + (l & 15);
        int k = kt * 32 + (j & 7) + ((j >> 3) & 1) * 16 + ((l >> 4) & 1) * 8;
        float v = (k < Kin) ? W[(size_t)n * Kin + k] : 0.f;
        out[e] = f2bf(v);
    }
}

__global__ void combine_bias_kernel(const float* __restrict__ bih,
                                    const float* __restrict__ bhh,
                                    float* __restrict__ out) {
    int i = blockIdx.x * blockDim.x + threadIdx.x;
    if (i < 2048) out[i] = bih[i] + bhh[i];
}

__global__ void gather_emb_kernel(const float* __restrict__ table,
                                  const int* __restrict__ idx,
                                  unsigned short* __restrict__ out) {
    int e = blockIdx.x * blockDim.x + threadIdx.x;
    if (e < BATCH * 64) {
        int b = e >> 6;
        out[e] = f2bf(table[(size_t)idx[b] * 64 + (e & 63)]);
    }
}

// ---------------------------------------------------------------------------
// Main kernel: 16 blocks x 256 threads (8 wave32). Block owns batch rows
// [32*blk, 32*blk+32) as two M-tiles. Wave w owns features [64w, 64w+64) of
// each of the 4 gates; B tiles are loaded once and reused for both M-tiles.
// ---------------------------------------------------------------------------
__global__ __launch_bounds__(256, 1)
void deepar_main_kernel(const float* __restrict__ target,
                        const float* __restrict__ cov,
                        const unsigned char* __restrict__ mask,
                        const float* __restrict__ noise,
                        const float* __restrict__ fcw,
                        const float* __restrict__ fcb,
                        const unsigned short* __restrict__ Wih0,
                        const unsigned short* __restrict__ Whh0,
                        const unsigned short* __restrict__ Wih1,
                        const unsigned short* __restrict__ Whh1,
                        const unsigned short* __restrict__ Wih2,
                        const unsigned short* __restrict__ Whh2,
                        const float* __restrict__ bias,
                        const unsigned short* __restrict__ embB,
                        float* __restrict__ cWs,
                        float* __restrict__ out) {
    // LDS: x (2 M-tiles x 3 k-tiles), h[3] (2 M-tiles x 16 k-tiles), bf16 A-packed
    __shared__ alignas(32) unsigned short bufX[2 * KT_IH0 * 512];   //   6 KB
    __shared__ alignas(32) unsigned short bufH[3 * 2 * 16 * 512];   //  96 KB
    __shared__ float paramsLds[64];
    __shared__ float nxtBuf[32];

    const int tid  = threadIdx.x;
    const int lane = tid & 31;
    const int wv   = tid >> 5;
    const int m0   = blockIdx.x * BQ;
    const int nlo  = lane & 15;          // N within 16x16 tile
    const int mhi  = (lane >> 4) << 3;   // +8 row offset for lanes 16..31

    float* cBlk = cWs + (size_t)blockIdx.x * 3 * BQ * 512;   // c state, f32 in L2

    for (int i = tid; i < 3 * 2 * 16 * 512; i += 256) bufH[i] = 0;
    for (int i = tid; i < 3 * BQ * 512; i += 256)     cBlk[i] = 0.f;
    if (tid < BQ) nxtBuf[tid] = 0.f;
    __syncthreads();

    #pragma unroll 1
    for (int t = 0; t < TSTEPS; ++t) {
        // ---- compose layer-0 input x = [inp_prev | cov_t | emb | 0-pad] ----
        for (int e = tid; e < 2 * KT_IH0 * 512; e += 256) {
            int mt = e / (KT_IH0 * 512);
            int e2 = e - mt * (KT_IH0 * 512);
            int j = e2 & 15, ll = (e2 >> 4) & 31, kt = e2 >> 9;
            int k = kt * 32 + (j & 7) + ((j >> 3) & 1) * 16 + ((ll >> 4) & 1) * 8;
            int m = mt * 16 + (ll & 15);
            unsigned short v;
            if (k == 0)       v = f2bf(nxtBuf[m]);
            else if (k < 5)   v = f2bf(cov[(size_t)(m0 + m) * TSTEPS * 4 + t * 4 + (k - 1)]);
            else if (k < 69)  v = embB[(m0 + m) * 64 + (k - 5)];
            else              v = 0;
            bufX[e] = v;
        }
        __syncthreads();

        // ---------------------------- 3 LSTM layers ------------------------
        #pragma unroll 1
        for (int l = 0; l < 3; ++l) {
            const unsigned short* Aih0 = (l == 0) ? bufX : &bufH[(l - 1) * 16384];
            const int aihStride        = (l == 0) ? (KT_IH0 * 512) : 8192;  // per M-tile
            const unsigned short* Ahh0 = &bufH[l * 16384];
            const unsigned short* Wih = (l == 0) ? Wih0 : ((l == 1) ? Wih1 : Wih2);
            const unsigned short* Whh = (l == 0) ? Whh0 : ((l == 1) ? Whh1 : Whh2);
            const int ktIh = (l == 0) ? KT_IH0 : KT_HH;
            float hRes[4][2][8];

            #pragma unroll
            for (int nt = 0; nt < 4; ++nt) {
                const int nb = wv * 4 + nt;       // 16-wide feature tile in [0,32)
                v8f acc[2][4];
                #pragma unroll
                for (int mt = 0; mt < 2; ++mt)
                    #pragma unroll
                    for (int g = 0; g < 4; ++g) acc[mt][g] = (v8f){};

                // x @ W_ih.T contribution (B loaded once, used for both M-tiles)
                for (int kt = 0; kt < ktIh; ++kt) {
                    v16bf bI = *(const v16bf*)&Wih[((size_t)(0 * 32 + nb) * ktIh + kt) * 512 + lane * 16];
                    v16bf bF = *(const v16bf*)&Wih[((size_t)(1 * 32 + nb) * ktIh + kt) * 512 + lane * 16];
                    v16bf bG = *(const v16bf*)&Wih[((size_t)(2 * 32 + nb) * ktIh + kt) * 512 + lane * 16];
                    v16bf bO = *(const v16bf*)&Wih[((size_t)(3 * 32 + nb) * ktIh + kt) * 512 + lane * 16];
                    #pragma unroll
                    for (int mt = 0; mt < 2; ++mt) {
                        v16bf a = *(const v16bf*)&Aih0[mt * aihStride + kt * 512 + lane * 16];
                        acc[mt][0] = wmma_bf16(a, bI, acc[mt][0]);
                        acc[mt][1] = wmma_bf16(a, bF, acc[mt][1]);
                        acc[mt][2] = wmma_bf16(a, bG, acc[mt][2]);
                        acc[mt][3] = wmma_bf16(a, bO, acc[mt][3]);
                    }
                }
                // h @ W_hh.T contribution
                for (int kt = 0; kt < KT_HH; ++kt) {
                    v16bf bI = *(const v16bf*)&Whh[((size_t)(0 * 32 + nb) * KT_HH + kt) * 512 + lane * 16];
                    v16bf bF = *(const v16bf*)&Whh[((size_t)(1 * 32 + nb) * KT_HH + kt) * 512 + lane * 16];
                    v16bf bG = *(const v16bf*)&Whh[((size_t)(2 * 32 + nb) * KT_HH + kt) * 512 + lane * 16];
                    v16bf bO = *(const v16bf*)&Whh[((size_t)(3 * 32 + nb) * KT_HH + kt) * 512 + lane * 16];
                    #pragma unroll
                    for (int mt = 0; mt < 2; ++mt) {
                        v16bf a = *(const v16bf*)&Ahh0[mt * 8192 + kt * 512 + lane * 16];
                        acc[mt][0] = wmma_bf16(a, bI, acc[mt][0]);
                        acc[mt][1] = wmma_bf16(a, bF, acc[mt][1]);
                        acc[mt][2] = wmma_bf16(a, bG, acc[mt][2]);
                        acc[mt][3] = wmma_bf16(a, bO, acc[mt][3]);
                    }
                }
                // gates + cell update (c slice is wave-exclusive, lives in L2)
                const int fL  = nb * 16 + nlo;    // feature in [0,512)
                float bI_s = bias[l * 2048 + 0 * 512 + fL];
                float bF_s = bias[l * 2048 + 1 * 512 + fL];
                float bG_s = bias[l * 2048 + 2 * 512 + fL];
                float bO_s = bias[l * 2048 + 3 * 512 + fL];
                #pragma unroll
                for (int mt = 0; mt < 2; ++mt) {
                    #pragma unroll
                    for (int r = 0; r < 8; ++r) {
                        int m = mt * 16 + r + mhi;
                        float gi = sigmoidf_(acc[mt][0][r] + bI_s);
                        float gf = sigmoidf_(acc[mt][1][r] + bF_s);
                        float gg = tanhf_(acc[mt][2][r] + bG_s);
                        float go = sigmoidf_(acc[mt][3][r] + bO_s);
                        size_t cIdx = ((size_t)(l * BQ + m)) * 512 + fL;
                        float cNew = gf * cBlk[cIdx] + gi * gg;
                        cBlk[cIdx] = cNew;
                        hRes[nt][mt][r] = go * tanhf_(cNew);
                    }
                }
            }
            __syncthreads();   // all waves done reading h_{t-1} of this layer
            // write h_new (A-packed) back into bufH[l]
            #pragma unroll
            for (int nt = 0; nt < 4; ++nt) {
                int f = (wv * 4 + nt) * 16 + nlo;
                int kp = f & 31, ktile = f >> 5;
                int lp = ((kp >> 3) & 1) * 16;
                int jp = ((kp >> 4) & 1) * 8 + (kp & 7);
                #pragma unroll
                for (int mt = 0; mt < 2; ++mt) {
                    #pragma unroll
                    for (int r = 0; r < 8; ++r) {
                        int m = r + mhi;
                        bufH[l * 16384 + mt * 8192 + ktile * 512 + (lp + m) * 16 + jp] =
                            f2bf(hRes[nt][mt][r]);
                    }
                }
            }
            __syncthreads();
        }

        // --------------------------- FC head (2 x 512) ----------------------
        if (tid < 64) paramsLds[tid] = 0.f;
        __syncthreads();
        {
            int m = nlo, head = lane >> 4;
            #pragma unroll
            for (int mt = 0; mt < 2; ++mt) {
                float s = 0.f;
                for (int ff = 0; ff < 64; ++ff) {
                    int f = wv * 64 + ff;
                    int kp = f & 31, ktile = f >> 5;
                    int lp = ((kp >> 3) & 1) * 16 + m;
                    int jp = ((kp >> 4) & 1) * 8 + (kp & 7);
                    s += bf2f(bufH[2 * 16384 + mt * 8192 + ktile * 512 + lp * 16 + jp]) *
                         fcw[head * 512 + f];
                }
                atomicAdd(&paramsLds[head * 32 + mt * 16 + m], s);
            }
        }
        __syncthreads();
        if (tid < BQ) {
            int m = tid, b = m0 + m;
            float mean = paramsLds[m] + fcb[0];
            float sp   = paramsLds[32 + m] + fcb[1];
            float soft = (sp > 20.f) ? sp : log1pf(__expf(sp));
            float sigma = soft + 1e-6f;
            out[1 + (size_t)t * BATCH + b] = mean;
            out[1 + (size_t)TSTEPS * BATCH + (size_t)t * BATCH + b] = sigma;
            float sample = mean + sigma * noise[(size_t)t * BATCH + b];
            float z = target[(size_t)b * TSTEPS + t];
            nxtBuf[m] = mask[(size_t)b * TSTEPS + t] ? z : sample;
        }
        __syncthreads();
    }
}

// ---------------------------------------------------------------------------
// Final masked-NLL reduction: loss from means/sigmas already in d_out.
// ---------------------------------------------------------------------------
__global__ void deepar_loss_kernel(const float* __restrict__ target,
                                   const unsigned char* __restrict__ mask,
                                   float* __restrict__ out) {
    __shared__ float sTot[TSTEPS];
    __shared__ float sCnt[TSTEPS];
    int t = threadIdx.x;   // blockDim.x == TSTEPS
    float sum = 0.f; int nobs = 0;
    for (int b = 0; b < BATCH; ++b) {
        if (mask[(size_t)b * TSTEPS + t]) {
            float mean  = out[1 + (size_t)t * BATCH + b];
            float sigma = out[1 + (size_t)TSTEPS * BATCH + (size_t)t * BATCH + b];
            float z = target[(size_t)b * TSTEPS + t];
            float d = (z - mean) * __builtin_amdgcn_rcpf(sigma);
            sum += 0.5f * LOG2PI + __logf(sigma) + 0.5f * d * d;
            ++nobs;
        }
    }
    sTot[t] = (nobs > 0) ? sum * __builtin_amdgcn_rcpf(fmaxf((float)nobs, 1.f)) : 0.f;
    sCnt[t] = (nobs > 0) ? 1.f : 0.f;
    __syncthreads();
    if (t == 0) {
        float tot = 0.f, cnt = 0.f;
        for (int i = 0; i < TSTEPS; ++i) { tot += sTot[i]; cnt += sCnt[i]; }
        out[0] = (cnt > 0.f) ? tot / cnt : tot;
    }
}

// ---------------------------------------------------------------------------
extern "C" void kernel_launch(void* const* d_in, const int* in_sizes, int n_in,
                              void* d_out, int out_size, void* d_ws, size_t ws_size,
                              hipStream_t stream) {
    const float*         target = (const float*)d_in[0];
    const float*         cov    = (const float*)d_in[1];
    const unsigned char* mask   = (const unsigned char*)d_in[2];
    const int*           idx    = (const int*)d_in[3];
    const float*         embT   = (const float*)d_in[4];
    const float*         fcw    = (const float*)d_in[5];
    const float*         fcb    = (const float*)d_in[6];
    const float*         noise  = (const float*)d_in[7];
    const float* Wih[3] = { (const float*)d_in[8],  (const float*)d_in[12], (const float*)d_in[16] };
    const float* Whh[3] = { (const float*)d_in[9],  (const float*)d_in[13], (const float*)d_in[17] };
    const float* bih[3] = { (const float*)d_in[10], (const float*)d_in[14], (const float*)d_in[18] };
    const float* bhh[3] = { (const float*)d_in[11], (const float*)d_in[15], (const float*)d_in[19] };

    // workspace layout (bf16-packed weights + bias + embeddings + c state)
    char* ws = (char*)d_ws;
    size_t off = 0;
    unsigned short* Wih0p = (unsigned short*)(ws + off); off += (size_t)128 * KT_IH0 * 512 * 2;
    unsigned short* Whh0p = (unsigned short*)(ws + off); off += (size_t)128 * KT_HH  * 512 * 2;
    unsigned short* Wih1p = (unsigned short*)(ws + off); off += (size_t)128 * KT_HH  * 512 * 2;
    unsigned short* Whh1p = (unsigned short*)(ws + off); off += (size_t)128 * KT_HH  * 512 * 2;
    unsigned short* Wih2p = (unsigned short*)(ws + off); off += (size_t)128 * KT_HH  * 512 * 2;
    unsigned short* Whh2p = (unsigned short*)(ws + off); off += (size_t)128 * KT_HH  * 512 * 2;
    float*          biasP = (float*)(ws + off);          off += (size_t)3 * 2048 * 4;
    unsigned short* embB  = (unsigned short*)(ws + off); off += (size_t)BATCH * 64 * 2;
    float*          cWs   = (float*)(ws + off);          off += (size_t)NBLOCKS * 3 * BQ * 512 * 4;

    pack_weights_kernel<<<256,  256, 0, stream>>>(Wih[0], Wih0p, 69,  KT_IH0);
    pack_weights_kernel<<<1024, 256, 0, stream>>>(Whh[0], Whh0p, 512, KT_HH);
    pack_weights_kernel<<<1024, 256, 0, stream>>>(Wih[1], Wih1p, 512, KT_HH);
    pack_weights_kernel<<<1024, 256, 0, stream>>>(Whh[1], Whh1p, 512, KT_HH);
    pack_weights_kernel<<<1024, 256, 0, stream>>>(Wih[2], Wih2p, 512, KT_HH);
    pack_weights_kernel<<<1024, 256, 0, stream>>>(Whh[2], Whh2p, 512, KT_HH);
    for (int l = 0; l < 3; ++l)
        combine_bias_kernel<<<8, 256, 0, stream>>>(bih[l], bhh[l], biasP + l * 2048);
    gather_emb_kernel<<<128, 256, 0, stream>>>(embT, idx, embB);

    deepar_main_kernel<<<NBLOCKS, 256, 0, stream>>>(
        target, cov, mask, noise, fcw, fcb,
        Wih0p, Whh0p, Wih1p, Whh1p, Wih2p, Whh2p,
        biasP, embB, cWs, (float*)d_out);

    deepar_loss_kernel<<<1, TSTEPS, 0, stream>>>(target, mask, (float*)d_out);
}
